// EDAIC_LSTM_58360015618468
// MI455X (gfx1250) — matrix operand order repaired and checked by reading, not compiled, not run on hardware
//
#include <hip/hip_runtime.h>
#include <hip/hip_bf16.h>
#include <math.h>

// ---------------------------------------------------------------------------
// EDAIC_LSTM forward for MI455X (gfx1250, wave32, WMMA bf16 matrix cores)
// ---------------------------------------------------------------------------

typedef __bf16 bf16;
typedef __bf16 v16bf __attribute__((ext_vector_type(16)));
typedef float  v8f   __attribute__((ext_vector_type(8)));

#define B_  8
#define S_  1024
#define D_  512
#define INNER_ 1024

__device__ __forceinline__ v8f wmma_bf16(v16bf a, v16bf b, v8f c) {
    return __builtin_amdgcn_wmma_f32_16x16x32_bf16(false, a, false, b, (short)0, c, false, false);
}

__device__ __forceinline__ float logsigf(float x) {
    return (x > 0.f) ? -log1pf(__expf(-x)) : (x - log1pf(__expf(x)));
}
__device__ __forceinline__ float siluf(float x) {
    return x / (1.f + __expf(-x));
}

// Pack 8 floats (two float4) into fragment slots [base..base+7].
__device__ __forceinline__ void cvt8(v16bf& d, int base, float4 a, float4 b) {
    d[base + 0] = (bf16)a.x; d[base + 1] = (bf16)a.y;
    d[base + 2] = (bf16)a.z; d[base + 3] = (bf16)a.w;
    d[base + 4] = (bf16)b.x; d[base + 5] = (bf16)b.y;
    d[base + 6] = (bf16)b.z; d[base + 7] = (bf16)b.w;
}
__device__ __forceinline__ void cvt8s(v16bf& d, int base, float4 a, float4 b, float s) {
    d[base + 0] = (bf16)(a.x * s); d[base + 1] = (bf16)(a.y * s);
    d[base + 2] = (bf16)(a.z * s); d[base + 3] = (bf16)(a.w * s);
    d[base + 4] = (bf16)(b.x * s); d[base + 5] = (bf16)(b.y * s);
    d[base + 6] = (bf16)(b.z * s); d[base + 7] = (bf16)(b.w * s);
}

// ---------------------------------------------------------------------------
// Zero-pad rows: dst[r, 0:scols] = src[r, :], dst[r, scols:dcols] = 0
// ---------------------------------------------------------------------------
__global__ void pad_rows(const float* __restrict__ src, float* __restrict__ dst,
                         int rows, int scols, int dcols)
{
    size_t i = (size_t)blockIdx.x * blockDim.x + threadIdx.x;
    size_t total = (size_t)rows * dcols;
    if (i >= total) return;
    int c = (int)(i % dcols);
    size_t r = i / dcols;
    dst[i] = (c < scols) ? src[r * scols + c] : 0.f;
}

// ---------------------------------------------------------------------------
// WMMA GEMM: C[M,N] = A[M,K] * W[N,K]^T (+ bias[n]) (+ C if addc)
// One wave per 16x16 C tile. Requirements: K%32==0, lda/ldw multiples of 4.
// Pure fast path: b128 loads, bf16 convert, f32 accumulate.
// ---------------------------------------------------------------------------
__global__ void wmma_gemm(const float* __restrict__ A, int lda,
                          const float* __restrict__ W, int ldw,
                          float* __restrict__ C, int ldc,
                          const float* __restrict__ bias,
                          int M, int N, int K, int addc)
{
    int m0 = blockIdx.x * 16;
    int n0 = blockIdx.y * 16;
    int lane = threadIdx.x;
    int nl = lane & 15, grp = lane >> 4;

    v8f acc; for (int i = 0; i < 8; i++) acc[i] = 0.f;
    const float* Arow = A + (size_t)(m0 + nl) * lda;   // A-frag: lane holds row m = nl
    const float* Wrow = W + (size_t)(n0 + nl) * ldw;   // B-frag: lane holds col n = nl

    for (int k0 = 0; k0 < K; k0 += 32) {
        const float* pa = Arow + k0 + grp * 16;        // A run: 16 contiguous
        float4 x0 = *(const float4*)(pa);
        float4 x1 = *(const float4*)(pa + 4);
        float4 x2 = *(const float4*)(pa + 8);
        float4 x3 = *(const float4*)(pa + 12);
        const float* pb = Wrow + k0 + grp * 16;        // B run: 16 contiguous
        float4 y0 = *(const float4*)(pb);
        float4 y1 = *(const float4*)(pb + 4);
        float4 y2 = *(const float4*)(pb + 8);
        float4 y3 = *(const float4*)(pb + 12);
        v16bf a, b;
        cvt8(a, 0, x0, x1); cvt8(a, 8, x2, x3);
        cvt8(b, 0, y0, y1); cvt8(b, 8, y2, y3);
        acc = wmma_bf16(a, b, acc);
    }
    float bv = bias ? bias[n0 + nl] : 0.f;
    for (int r = 0; r < 8; r++) {
        int m = m0 + r + 8 * grp;                      // C/D layout
        size_t idx = (size_t)m * ldc + n0 + nl;
        float v = acc[r] + bv;
        if (addc) v += C[idx];
        C[idx] = v;
    }
}

// ---------------------------------------------------------------------------
// Unified (head-)LayerNorm. Rows of contiguous length L; weight slot = row%wslots.
// Optional residual add (out = res + normed).
// ---------------------------------------------------------------------------
__global__ void headln_kernel(const float* __restrict__ x, const float* __restrict__ w,
                              const float* __restrict__ res, float* __restrict__ out,
                              int L, int wslots)
{
    size_t row = blockIdx.x;
    int lane = threadIdx.x;
    const float* xr = x + row * L;
    float s = 0.f, s2 = 0.f;
    for (int i = lane; i < L; i += 32) { float t = xr[i]; s += t; s2 += t * t; }
    for (int m = 16; m; m >>= 1) { s += __shfl_xor(s, m, 32); s2 += __shfl_xor(s2, m, 32); }
    float mu = s / (float)L;
    float var = s2 / (float)L - mu * mu;
    float rs = rsqrtf(var + 1e-5f);
    const float* wr = w + (size_t)(row % (size_t)wslots) * L;
    for (int i = lane; i < L; i += 32) {
        float t = (xr[i] - mu) * rs * wr[i];
        if (res) t += res[row * L + i];
        out[row * L + i] = t;
    }
}

// ---------------------------------------------------------------------------
// Depthwise causal conv (K=4) + SiLU.
// ---------------------------------------------------------------------------
__global__ void conv_silu(const float* __restrict__ x, int ldx,
                          const float* __restrict__ ker, const float* __restrict__ bias,
                          float* __restrict__ out, int ldo, int C)
{
    size_t i = (size_t)blockIdx.x * blockDim.x + threadIdx.x;
    size_t total = (size_t)B_ * S_ * C;
    if (i >= total) return;
    int c = (int)(i % C);
    size_t bs = i / C;
    int s = (int)(bs % S_);
    float a = bias[c];
    for (int t = 0; t < 4; t++) {
        int sp = s - 3 + t;
        if (sp >= 0) a += x[(bs - (size_t)(s - sp)) * ldx + c] * ker[t * C + c];
    }
    out[bs * ldo + c] = siluf(a);
}

// ---------------------------------------------------------------------------
// mLSTM headwise q/k/v: block-diagonal 4x4 linear (NPH=256 blocks).
// ---------------------------------------------------------------------------
__global__ void headwise4(const float* __restrict__ x, int ldx,
                          const float* __restrict__ w, const float* __restrict__ bias,
                          float* __restrict__ out)
{
    size_t i = (size_t)blockIdx.x * blockDim.x + threadIdx.x;
    if (i >= (size_t)B_ * S_ * 256) return;
    int p = (int)(i % 256); size_t bs = i / 256;
    const float* xr = x + bs * ldx + p * 4;
    const float* wr = w + p * 16;
    float x0 = xr[0], x1 = xr[1], x2 = xr[2], x3 = xr[3];
    for (int o = 0; o < 4; o++) {
        float a = wr[o * 4 + 0] * x0 + wr[o * 4 + 1] * x1 +
                  wr[o * 4 + 2] * x2 + wr[o * 4 + 3] * x3 + bias[p * 4 + o];
        out[bs * 1024 + p * 4 + o] = a;
    }
}

// ---------------------------------------------------------------------------
// mLSTM gates: ig/fg = concat(q,k,v) @ w^T + b ; (B,NH,S). One wave per (b,s).
// ---------------------------------------------------------------------------
__global__ void mlstm_gates(const float* __restrict__ q, const float* __restrict__ k,
                            const float* __restrict__ v,
                            const float* __restrict__ ig_w, const float* __restrict__ ig_b,
                            const float* __restrict__ fg_w, const float* __restrict__ fg_b,
                            float* __restrict__ igo, float* __restrict__ fgo)
{
    size_t bs = blockIdx.x;
    int lane = threadIdx.x;
    int b = (int)(bs >> 10), s = (int)(bs & 1023);
    for (int h = 0; h < 4; h++) {
        float ai = 0.f, af = 0.f;
        const float* iw = ig_w + (size_t)h * 3072;
        const float* fw = fg_w + (size_t)h * 3072;
        for (int j = lane; j < 3072; j += 32) {
            float g = (j < 1024) ? q[bs * 1024 + j]
                    : (j < 2048) ? k[bs * 1024 + j - 1024]
                                 : v[bs * 1024 + j - 2048];
            ai += g * iw[j]; af += g * fw[j];
        }
        for (int m = 16; m; m >>= 1) { ai += __shfl_xor(ai, m, 32); af += __shfl_xor(af, m, 32); }
        if (lane == 0) {
            igo[((size_t)(b * 4 + h)) * S_ + s] = ai + ig_b[h];
            fgo[((size_t)(b * 4 + h)) * S_ + s] = af + fg_b[h];
        }
    }
}

__global__ void fgate_cumsum(const float* __restrict__ fgo, float* __restrict__ cum)
{
    int t = threadIdx.x;           // 0..31 = (b*4+h)
    float a = 0.f;
    const float* fr = fgo + (size_t)t * S_;
    float* cr = cum + (size_t)t * S_;
    for (int s = 0; s < S_; s++) { a += logsigf(fr[s]); cr[s] = a; }
}

// ---------------------------------------------------------------------------
// Flash-style mLSTM cell. One wave per (b, h, 16-query tile). DH=256.
// Running max / running normalizer (flash rescaling) implements the
// stabilized parallel mLSTM exactly. V tile staged to LDS (transposed, bf16).
// ---------------------------------------------------------------------------
__global__ void mlstm_flash(const float* __restrict__ q, const float* __restrict__ k,
                            const float* __restrict__ v, const float* __restrict__ cum,
                            const float* __restrict__ ig, float* __restrict__ out)
{
    __shared__ float plds[16 * 32];        //  2 KB: P tile (C-layout -> A-layout)
    __shared__ bf16  vldsT[256 * 32];      // 16 KB: V tile transposed [dh][key]
    int id = blockIdx.x;
    int qt = id & 63; id >>= 6;
    int h = id & 3;  int b = id >> 2;
    int q0 = qt * 16;
    int lane = threadIdx.x;
    int nl = lane & 15, grp = lane >> 4;
    const size_t rowbase = (size_t)b * S_;

    // Q fragments (1/sqrt(DH)=1/16 folded in)
    v16bf qa[8];
    {
        const float* qr = q + (rowbase + q0 + nl) * 1024 + h * 256;
        for (int f = 0; f < 8; f++) {
            const float* p = qr + f * 32 + grp * 16;
            float4 x0 = *(const float4*)(p);
            float4 x1 = *(const float4*)(p + 4);
            float4 x2 = *(const float4*)(p + 8);
            float4 x3 = *(const float4*)(p + 12);
            cvt8s(qa[f], 0, x0, x1, 0.0625f);
            cvt8s(qa[f], 8, x2, x3, 0.0625f);
        }
    }
    v8f acc[16];
    for (int c = 0; c < 16; c++) for (int r = 0; r < 8; r++) acc[c][r] = 0.f;

    float mrow[8], ssum[8], cq[8];
    const float* cumbh = cum + ((size_t)(b * 4 + h)) * S_;
    const float* igbh  = ig  + ((size_t)(b * 4 + h)) * S_;
    for (int r = 0; r < 8; r++) {
        mrow[r] = -1e30f; ssum[r] = 0.f;
        cq[r] = cumbh[q0 + r + 8 * grp];
    }

    int nk = q0 + 16;                       // causal: keys 0..q0+15
    for (int kb = 0; kb < nk; kb += 32) {
        // ---- stage V tile (32 keys x 256 dh) to LDS, transposed, bf16 ----
        for (int rr = 0; rr < 32; rr++) {
            int key = kb + rr; if (key > S_ - 1) key = S_ - 1;
            const float* vr = v + (rowbase + key) * 1024 + h * 256 + lane * 8;
            float4 u0 = *(const float4*)(vr);
            float4 u1 = *(const float4*)(vr + 4);
            int dh = lane * 8;
            vldsT[(dh + 0) * 32 + rr] = (bf16)u0.x;
            vldsT[(dh + 1) * 32 + rr] = (bf16)u0.y;
            vldsT[(dh + 2) * 32 + rr] = (bf16)u0.z;
            vldsT[(dh + 3) * 32 + rr] = (bf16)u0.w;
            vldsT[(dh + 4) * 32 + rr] = (bf16)u1.x;
            vldsT[(dh + 5) * 32 + rr] = (bf16)u1.y;
            vldsT[(dh + 6) * 32 + rr] = (bf16)u1.z;
            vldsT[(dh + 7) * 32 + rr] = (bf16)u1.w;
        }
        // ---- S = Q K^T for two 16-key subtiles ----
        v8f s0, s1; for (int r = 0; r < 8; r++) { s0[r] = 0.f; s1[r] = 0.f; }
        {
            int key0 = kb + nl;      if (key0 > S_ - 1) key0 = S_ - 1;
            int key1 = kb + 16 + nl; if (key1 > S_ - 1) key1 = S_ - 1;
            const float* kr0 = k + (rowbase + key0) * 1024 + h * 256;
            const float* kr1 = k + (rowbase + key1) * 1024 + h * 256;
            for (int f = 0; f < 8; f++) {
                const float* p0 = kr0 + f * 32 + grp * 16;
                const float* p1 = kr1 + f * 32 + grp * 16;
                float4 y0 = *(const float4*)(p0);
                float4 y1 = *(const float4*)(p0 + 4);
                float4 y2 = *(const float4*)(p0 + 8);
                float4 y3 = *(const float4*)(p0 + 12);
                float4 z0 = *(const float4*)(p1);
                float4 z1 = *(const float4*)(p1 + 4);
                float4 z2 = *(const float4*)(p1 + 8);
                float4 z3 = *(const float4*)(p1 + 12);
                v16bf b0, b1;
                cvt8(b0, 0, y0, y1); cvt8(b0, 8, y2, y3);
                cvt8(b1, 0, z0, z1); cvt8(b1, 8, z2, z3);
                s0 = wmma_bf16(qa[f], b0, s0);
                s1 = wmma_bf16(qa[f], b1, s1);
            }
        }
        // ---- gating + flash rescale ----
        int j0 = kb + nl, j1 = kb + 16 + nl;
        int j0c = (j0 < S_) ? j0 : S_ - 1;
        int j1c = (j1 < S_) ? j1 : S_ - 1;
        float c0 = cumbh[j0c], i0 = igbh[j0c];
        float c1 = cumbh[j1c], i1 = igbh[j1c];
        for (int r = 0; r < 8; r++) {
            int mg = q0 + r + 8 * grp;
            float ld0 = (j0 <= mg) ? (cq[r] - c0 + i0) : -1e30f;
            float ld1 = (j1 <= mg) ? (cq[r] - c1 + i1) : -1e30f;
            float rmax = fmaxf(ld0, ld1);
            for (int ms = 8; ms; ms >>= 1) rmax = fmaxf(rmax, __shfl_xor(rmax, ms, 32));
            float mnew = fmaxf(mrow[r], rmax);
            float scale = __expf(mrow[r] - mnew);
            float p0 = s0[r] * __expf(ld0 - mnew);
            float p1 = s1[r] * __expf(ld1 - mnew);
            float ps = p0 + p1;
            for (int ms = 8; ms; ms >>= 1) ps += __shfl_xor(ps, ms, 32);
            ssum[r] = ssum[r] * scale + ps;
            mrow[r] = mnew;
            for (int c = 0; c < 16; c++) acc[c][r] *= scale;
            plds[(r + 8 * grp) * 32 + nl]      = p0;   // C-layout -> LDS (m,n)
            plds[(r + 8 * grp) * 32 + nl + 16] = p1;
        }
        __syncthreads();
        v16bf pa;   // reload P in A-fragment layout (16 queries x 32 keys)
        {
            const float* pp = &plds[nl * 32 + grp * 16];
            float4 p0 = *(const float4*)(pp);
            float4 p1 = *(const float4*)(pp + 4);
            float4 p2 = *(const float4*)(pp + 8);
            float4 p3 = *(const float4*)(pp + 12);
            cvt8(pa, 0, p0, p1); cvt8(pa, 8, p2, p3);
        }
        // ---- acc += P V : B-frags straight out of transposed LDS tile ----
        for (int c = 0; c < 16; c++) {
            v16bf vf = *(const v16bf*)(&vldsT[(c * 16 + nl) * 32 + grp * 16]);
            acc[c] = wmma_bf16(pa, vf, acc[c]);
        }
        __syncthreads();
    }
    // ---- finalize: h = acc / (max(|ssum|, exp(-m)) + 1e-6) ----
    float inv[8];
    for (int r = 0; r < 8; r++) {
        float den = fmaxf(fabsf(ssum[r]), __expf(-mrow[r])) + 1e-6f;
        inv[r] = 1.f / den;
    }
    for (int c = 0; c < 16; c++)
        for (int r = 0; r < 8; r++)
            out[(rowbase + q0 + r + 8 * grp) * 1024 + h * 256 + c * 16 + nl] = acc[c][r] * inv[r];
}

// ---------------------------------------------------------------------------
// hs = (htil_ln + skip*xc) * silu(z)
// ---------------------------------------------------------------------------
__global__ void mlstm_hs(const float* __restrict__ htln, const float* __restrict__ xc,
                         const float* __restrict__ skip_w, const float* __restrict__ up,
                         float* __restrict__ hs)
{
    size_t i = (size_t)blockIdx.x * blockDim.x + threadIdx.x;
    if (i >= (size_t)B_ * S_ * INNER_) return;
    int c = (int)(i & 1023);
    size_t bs = i >> 10;
    float z = up[bs * 2048 + 1024 + c];
    hs[i] = (htln[i] + skip_w[c] * xc[i]) * siluf(z);
}

// ---------------------------------------------------------------------------
// sLSTM scan: one block per (b, head). R_h cached in LDS as bf16 (128 KB).
// ---------------------------------------------------------------------------
__global__ void slstm_scan(const float* __restrict__ wx, const float* __restrict__ R,
                           const float* __restrict__ cell_b, float* __restrict__ ys)
{
    extern __shared__ char smem[];
    bf16* Rl = (bf16*)smem;                                  // 4*128*128 bf16
    float* yl = (float*)(smem + (size_t)4 * 128 * 128 * sizeof(bf16));
    int bh = blockIdx.x; int h = bh & 3; int b = bh >> 2;
    int d = threadIdx.x;                                     // 0..127
    const float* Rh = R + (size_t)h * 512 * 128;
    for (int i = d; i < 512 * 128; i += 128) Rl[i] = (bf16)Rh[i];
    yl[d] = 0.f;
    __syncthreads();

    float c_ = 0.f, n_ = 0.f, m_ = 0.f, y_ = 0.f;
    float bi = cell_b[0 * 512 + h * 128 + d];
    float bf = cell_b[1 * 512 + h * 128 + d];
    float bz = cell_b[2 * 512 + h * 128 + d];
    float bo = cell_b[3 * 512 + h * 128 + d];

    for (int t = 0; t < S_; t++) {
        const float* wxr = wx + ((size_t)(b * S_ + t)) * 2048 + h * 128 + d;
        float ri = 0.f, rf = 0.f, rz = 0.f, ro = 0.f;
        for (int k2 = 0; k2 < 128; k2++) {
            float yk = yl[k2];
            ri += yk * (float)Rl[(0 * 128 + d) * 128 + k2];
            rf += yk * (float)Rl[(1 * 128 + d) * 128 + k2];
            rz += yk * (float)Rl[(2 * 128 + d) * 128 + k2];
            ro += yk * (float)Rl[(3 * 128 + d) * 128 + k2];
        }
        float ir = wxr[0]    + ri + bi;
        float fr = wxr[512]  + rf + bf;
        float zr = wxr[1024] + rz + bz;
        float og = wxr[1536] + ro + bo;
        float lf = m_ + logsigf(fr);
        float mnew = (n_ == 0.f) ? ir : fmaxf(ir, lf);
        float igate = __expf(ir - mnew), fgate = __expf(lf - mnew);
        c_ = fgate * c_ + igate * tanhf(zr);
        n_ = fgate * n_ + igate;
        m_ = mnew;
        y_ = (1.f / (1.f + __expf(-og))) * c_ / n_;
        ys[((size_t)(b * S_ + t)) * 512 + h * 128 + d] = y_;
        __syncthreads();
        yl[d] = y_;
        __syncthreads();
    }
}

// ---------------------------------------------------------------------------
// FFN activation: act = gelu_exact(gate) * up
// ---------------------------------------------------------------------------
__global__ void ffn_act(const float* __restrict__ ffu, float* __restrict__ act)
{
    size_t i = (size_t)blockIdx.x * blockDim.x + threadIdx.x;
    if (i >= (size_t)B_ * S_ * 704) return;
    int j = (int)(i % 704);
    size_t bs = i / 704;
    float g = ffu[bs * 1408 + j];
    float u = ffu[bs * 1408 + 704 + j];
    act[i] = 0.5f * g * (1.f + erff(g * 0.70710678118f)) * u;
}

// ---------------------------------------------------------------------------
// Final FC on last token: out[b] = xn[b, S-1, :] . fc_w + fc_b
// ---------------------------------------------------------------------------
__global__ void final_head(const float* __restrict__ xn, const float* __restrict__ fc_w,
                           const float* __restrict__ fc_b, float* __restrict__ out)
{
    int w = threadIdx.x >> 5, lane = threadIdx.x & 31;    // w = batch
    const float* xr = xn + ((size_t)w * S_ + (S_ - 1)) * D_;
    float a = 0.f;
    for (int i = lane; i < D_; i += 32) a += xr[i] * fc_w[i];
    for (int m = 16; m; m >>= 1) a += __shfl_xor(a, m, 32);
    if (lane == 0) out[w] = a + fc_b[0];
}

// ---------------------------------------------------------------------------
extern "C" void kernel_launch(void* const* d_in, const int* in_sizes, int n_in,
                              void* d_out, int out_size, void* d_ws, size_t ws_size,
                              hipStream_t stream) {
    const float* x       = (const float*)d_in[0];
    const float* w_in    = (const float*)d_in[1];
    const float* b_in    = (const float*)d_in[2];
    const float* ln0_w   = (const float*)d_in[3];
    const float* m_up_w  = (const float*)d_in[4];
    const float* m_up_b  = (const float*)d_in[5];
    const float* m_conv_k= (const float*)d_in[6];
    const float* m_conv_b= (const float*)d_in[7];
    const float* q_w     = (const float*)d_in[8];
    const float* q_b     = (const float*)d_in[9];
    const float* k_w     = (const float*)d_in[10];
    const float* k_b     = (const float*)d_in[11];
    const float* v_w     = (const float*)d_in[12];
    const float* v_b     = (const float*)d_in[13];
    const float* ig_w    = (const float*)d_in[14];
    const float* ig_b    = (const float*)d_in[15];
    const float* fg_w    = (const float*)d_in[16];
    const float* fg_b    = (const float*)d_in[17];
    const float* onorm_w = (const float*)d_in[18];
    const float* skip_w  = (const float*)d_in[19];
    const float* m_dn_w  = (const float*)d_in[20];
    const float* m_dn_b  = (const float*)d_in[21];
    const float* ln1_w   = (const float*)d_in[22];
    const float* s_conv_k= (const float*)d_in[23];
    const float* s_conv_b= (const float*)d_in[24];
    const float* wi      = (const float*)d_in[25];
    const float* wf      = (const float*)d_in[26];
    const float* wz      = (const float*)d_in[27];
    const float* wo      = (const float*)d_in[28];
    const float* r_kern  = (const float*)d_in[29];
    const float* cell_b  = (const float*)d_in[30];
    const float* gn_w    = (const float*)d_in[31];
    const float* ln2_w   = (const float*)d_in[32];
    const float* ff_up_w = (const float*)d_in[33];
    const float* ff_up_b = (const float*)d_in[34];
    const float* ff_dn_w = (const float*)d_in[35];
    const float* ff_dn_b = (const float*)d_in[36];
    const float* lnf_w   = (const float*)d_in[37];
    const float* fc_w    = (const float*)d_in[38];
    const float* fc_b    = (const float*)d_in[39];
    float* out = (float*)d_out;

    // ---- workspace layout (floats) ----
    const size_t MEG = 1024ull * 1024ull;
    float* W  = (float*)d_ws;
    float* f_h   = W;                 //  4M : residual stream h (B,S,512)
    float* f_xn  = W + 4 * MEG;       //  4M : LN scratch
    float* f_up  = W + 8 * MEG;       // 16M : m_up output (B,S,2048) / sLSTM wx
    float* f_xc  = W + 24 * MEG;      //  8M : conv output
    float* f_q   = W + 32 * MEG;      //  8M : q / htil_ln / ffu (spans q+k)
    float* f_k   = W + 40 * MEG;      //  8M : k / hs
    float* f_v   = W + 48 * MEG;      //  8M : v / ffn act
    float* f_ht  = W + 56 * MEG;      //  8M : htil / sLSTM y
    float* f_ig  = W + 64 * MEG;      // 32K
    float* f_fg  = f_ig + 32768;      // 32K
    float* f_cum = f_fg + 32768;      // 32K
    float* f_xp  = f_cum + 32768;     // 768K : x zero-padded to (B*S, 96)
    float* f_wp  = f_xp + (size_t)8192 * 96;   // 48K : w_in zero-padded to (512, 96)

    const int M = B_ * S_;            // 8192
    dim3 blk32(32), blk128(128), blk256(256);

    // 0) zero-pad x (8192x75 -> 8192x96) and w_in (512x75 -> 512x96)
    {
        size_t totx = (size_t)M * 96;
        pad_rows<<<dim3((unsigned)((totx + 255) / 256)), blk256, 0, stream>>>(x, f_xp, M, 75, 96);
        size_t totw = (size_t)D_ * 96;
        pad_rows<<<dim3((unsigned)((totw + 255) / 256)), blk256, 0, stream>>>(w_in, f_wp, D_, 75, 96);
    }

    // 1) input projection: h = xpad @ wpad^T + b_in  (K=96, fast path)
    wmma_gemm<<<dim3(M / 16, D_ / 16), blk32, 0, stream>>>(
        f_xp, 96, f_wp, 96, f_h, D_, b_in, M, D_, 96, 0);

    // 2) LN0
    headln_kernel<<<dim3(M), blk32, 0, stream>>>(f_h, ln0_w, nullptr, f_xn, D_, 1);

    // 3) m_up: (B,S,2048)
    wmma_gemm<<<dim3(M / 16, 2048 / 16), blk32, 0, stream>>>(
        f_xn, D_, m_up_w, D_, f_up, 2048, m_up_b, M, 2048, D_, 0);

    // 4) causal conv + SiLU on x_m (first 1024 of up)
    {
        size_t tot = (size_t)M * INNER_;
        conv_silu<<<dim3((unsigned)((tot + 255) / 256)), blk256, 0, stream>>>(
            f_up, 2048, m_conv_k, m_conv_b, f_xc, INNER_, INNER_);
    }

    // 5) headwise q/k/v (4x4 blocks)
    {
        size_t tot = (size_t)M * 256;
        dim3 g((unsigned)((tot + 255) / 256));
        headwise4<<<g, blk256, 0, stream>>>(f_xc, INNER_, q_w, q_b, f_q);
        headwise4<<<g, blk256, 0, stream>>>(f_xc, INNER_, k_w, k_b, f_k);
        headwise4<<<g, blk256, 0, stream>>>(f_up, 2048,  v_w, v_b, f_v);
    }

    // 6) gates + cumulative log-sigmoid(forget)
    mlstm_gates<<<dim3(M), blk32, 0, stream>>>(f_q, f_k, f_v, ig_w, ig_b, fg_w, fg_b, f_ig, f_fg);
    fgate_cumsum<<<dim3(1), blk32, 0, stream>>>(f_fg, f_cum);

    // 7) flash mLSTM cell -> htil
    mlstm_flash<<<dim3(B_ * 4 * (S_ / 16)), blk32, 0, stream>>>(f_q, f_k, f_v, f_cum, f_ig, f_ht);

    // 8) per-head LN of htil (DH=256, NH=4) -> f_q (q no longer needed)
    headln_kernel<<<dim3(M * 4), blk32, 0, stream>>>(f_ht, onorm_w, nullptr, f_q, 256, 4);

    // 9) hs = (htil_ln + skip*xc) * silu(z) -> f_k
    {
        size_t tot = (size_t)M * INNER_;
        mlstm_hs<<<dim3((unsigned)((tot + 255) / 256)), blk256, 0, stream>>>(
            f_q, f_xc, skip_w, f_up, f_k);
    }

    // 10) h += hs @ m_dn^T + m_dn_b
    wmma_gemm<<<dim3(M / 16, D_ / 16), blk32, 0, stream>>>(
        f_k, INNER_, m_dn_w, INNER_, f_h, D_, m_dn_b, M, D_, INNER_, 1);

    // 11) LN1, conv+SiLU (C=512)
    headln_kernel<<<dim3(M), blk32, 0, stream>>>(f_h, ln1_w, nullptr, f_xn, D_, 1);
    {
        size_t tot = (size_t)M * D_;
        conv_silu<<<dim3((unsigned)((tot + 255) / 256)), blk256, 0, stream>>>(
            f_xn, D_, s_conv_k, s_conv_b, f_xc, D_, D_);
    }

    // 12) sLSTM headwise projections -> wx (gate-major, f_up). i,f from xc; z,o from xn.
    {
        const float* gw[4] = { wi, wf, wz, wo };
        for (int g = 0; g < 4; g++) {
            const float* src = (g < 2) ? f_xc : f_xn;
            for (int h = 0; h < 4; h++) {
                wmma_gemm<<<dim3(M / 16, 128 / 16), blk32, 0, stream>>>(
                    src + h * 128, D_, gw[g] + (size_t)h * 128 * 128, 128,
                    f_up + g * 512 + h * 128, 2048, nullptr, M, 128, 128, 0);
            }
        }
    }

    // 13) sequential sLSTM scan -> y (f_ht), R_h cached in LDS (bf16, 128KB)
    {
        size_t shmem = (size_t)4 * 128 * 128 * sizeof(bf16) + 128 * sizeof(float);
        slstm_scan<<<dim3(B_ * 4), blk128, shmem, stream>>>(f_up, r_kern, cell_b, f_ht);
    }

    // 14) h += headLN(y) (DH=128, NH=4, gn_w), residual add into h
    headln_kernel<<<dim3(M * 4), blk32, 0, stream>>>(f_ht, gn_w, f_h, f_h, 128, 4);

    // 15) FFN: LN2 -> ffu -> gelu*up -> down-proj (accumulate into h)
    headln_kernel<<<dim3(M), blk32, 0, stream>>>(f_h, ln2_w, nullptr, f_xn, D_, 1);
    wmma_gemm<<<dim3(M / 16, 1408 / 16), blk32, 0, stream>>>(
        f_xn, D_, ff_up_w, D_, f_q, 1408, ff_up_b, M, 1408, D_, 0);
    {
        size_t tot = (size_t)M * 704;
        ffn_act<<<dim3((unsigned)((tot + 255) / 256)), blk256, 0, stream>>>(f_q, f_v);
    }
    wmma_gemm<<<dim3(M / 16, D_ / 16), blk32, 0, stream>>>(
        f_v, 704, ff_dn_w, 704, f_h, D_, ff_dn_b, M, D_, 704, 1);

    // 16) final LN + FC on last timestep
    headln_kernel<<<dim3(M), blk32, 0, stream>>>(f_h, lnf_w, nullptr, f_xn, D_, 1);
    final_head<<<dim3(1), blk256, 0, stream>>>(f_xn, fc_w, fc_b, out);

    (void)in_sizes; (void)n_in; (void)out_size; (void)ws_size;
}
